// PAM_755914244319
// MI455X (gfx1250) — compile-verified, matrix operand order checked by reading, not run on hardware
//
#include <hip/hip_runtime.h>
#include <hip/hip_bf16.h>
#include <stdint.h>

#define BATCH 8
#define CH    64
#define CG    16
#define HH    128
#define WW    128
#define NROWS (BATCH*HH)          // 1024 attention rows
#define IMG_ELEMS ((size_t)BATCH*CH*HH*WW)   // 8,388,608
#define IMG_BYTES (IMG_ELEMS*4)

typedef __attribute__((ext_vector_type(16))) _Float16 v16h;
typedef __attribute__((ext_vector_type(8)))  float    v8f;

// ---------------------------------------------------------------------------
// BN statistics: mean + rstd per channel, per image (training-mode batch stats)
// ---------------------------------------------------------------------------
__global__ __launch_bounds__(256) void bn_stats_kernel(
    const float* __restrict__ xl, const float* __restrict__ xr,
    float* __restrict__ stats)
{
    int c = blockIdx.x, img = blockIdx.y;
    const float* x = img ? xr : xl;
    float s1 = 0.f, s2 = 0.f;
    const int NP = BATCH * HH * WW;          // 131072 per channel
    for (int i = threadIdx.x; i < NP; i += 256) {
        int bb = i >> 14, p = i & 16383;
        float v = x[((size_t)bb*CH + c)*HH*WW + p];
        s1 += v; s2 += v*v;
    }
    __shared__ float r1[256], r2[256];
    r1[threadIdx.x] = s1; r2[threadIdx.x] = s2; __syncthreads();
    for (int off = 128; off; off >>= 1) {
        if (threadIdx.x < off) { r1[threadIdx.x] += r1[threadIdx.x+off];
                                 r2[threadIdx.x] += r2[threadIdx.x+off]; }
        __syncthreads();
    }
    if (threadIdx.x == 0) {
        float mean = r1[0] / (float)NP;
        float var  = r2[0] / (float)NP - mean*mean;
        stats[img*CH + c]       = mean;
        stats[128 + img*CH + c] = rsqrtf(var + 1e-5f);
    }
}

// ---------------------------------------------------------------------------
// Grouped 3x3 conv, LDS-tiled.  phase 0: in = BN(xraw), out = LeakyReLU(conv+b)
//                               phase 1: in = y1,       out = conv+b + BN(xraw)
// ---------------------------------------------------------------------------
__global__ __launch_bounds__(256) void conv3x3_kernel(
    const float* __restrict__ xraw, const float* __restrict__ y1,
    const float* __restrict__ wgt,  const float* __restrict__ bias,
    const float* __restrict__ stats,
    const float* __restrict__ gamma, const float* __restrict__ beta,
    float* __restrict__ out, int img, int phase)
{
    __shared__ float lin[CG][10][34];
    __shared__ float lw[CG][CG][9];
    int tile = blockIdx.x;            // 64 tiles: 4 x-tiles(32) * 16 y-tiles(8)
    int g = blockIdx.y, bb = blockIdx.z;
    int xt = (tile & 3) * 32, yt = (tile >> 2) * 8;
    int tx = threadIdx.x & 31, ty = threadIdx.x >> 5;

    for (int i = threadIdx.x; i < CG*CG*9; i += 256) {
        int co = i / (CG*9), r = i % (CG*9);
        lw[co][r/9][r%9] = wgt[((size_t)(g*CG+co))*CG*9 + r];
    }
    for (int i = threadIdx.x; i < CG*10*34; i += 256) {
        int ci = i / 340, r = i % 340, yy = r / 34, xx = r % 34;
        int gy = yt + yy - 1, gx = xt + xx - 1;
        float v = 0.f;
        if (gy >= 0 && gy < HH && gx >= 0 && gx < WW) {
            int ch = g*CG + ci;
            size_t gi = (((size_t)bb*CH + ch)*HH + gy)*WW + gx;
            if (phase == 0) {
                float xv = xraw[gi];
                v = gamma[ch]*(xv - stats[img*CH+ch])*stats[128+img*CH+ch] + beta[ch];
            } else {
                v = y1[gi];
            }
        }
        lin[ci][yy][xx] = v;
    }
    __syncthreads();

    int gy = yt + ty, gx = xt + tx;
    for (int co = 0; co < CG; ++co) {
        float acc = bias[g*CG + co];
        for (int ci = 0; ci < CG; ++ci) {
            #pragma unroll
            for (int ky = 0; ky < 3; ++ky)
                #pragma unroll
                for (int kx = 0; kx < 3; ++kx)
                    acc += lw[co][ci][ky*3+kx] * lin[ci][ty+ky][tx+kx];
        }
        int ch = g*CG + co;
        size_t gi = (((size_t)bb*CH + ch)*HH + gy)*WW + gx;
        float res;
        if (phase == 0) {
            res = acc > 0.f ? acc : 0.1f*acc;
        } else {
            float xv = xraw[gi];
            float bn = gamma[ch]*(xv - stats[img*CH+ch])*stats[128+img*CH+ch] + beta[ch];
            res = acc + bn;
        }
        out[gi] = res;
    }
}

// ---------------------------------------------------------------------------
// 1x1 grouped conv + row-mean removal (axis=W), output f16 [n][w][c]
// ---------------------------------------------------------------------------
__global__ __launch_bounds__(128) void qk_kernel(
    const float* __restrict__ resb, const float* __restrict__ w11,
    const float* __restrict__ bias, _Float16* __restrict__ outh)
{
    int co = blockIdx.x, y = blockIdx.y, bb = blockIdx.z;
    int x = threadIdx.x;
    int g = co >> 4;
    float acc = bias[co];
    #pragma unroll
    for (int ci = 0; ci < CG; ++ci)
        acc += w11[co*CG + ci] * resb[(((size_t)bb*CH + g*CG + ci)*HH + y)*WW + x];
    __shared__ float red[128];
    red[x] = acc; __syncthreads();
    for (int off = 64; off; off >>= 1) {
        if (x < off) red[x] += red[x+off];
        __syncthreads();
    }
    float mean = red[0] * (1.f/128.f);
    outh[(((size_t)(bb*HH + y))*WW + x)*CH + co] = (_Float16)(acc - mean);
}

// ---------------------------------------------------------------------------
// f16 transposed copy:  [b][c][h][w] f32  ->  [n=b*h][c][w] f16
// ---------------------------------------------------------------------------
__global__ __launch_bounds__(256) void tof16t_kernel(
    const float* __restrict__ x, _Float16* __restrict__ outh)
{
    size_t idx = (size_t)blockIdx.x*256 + threadIdx.x;
    if (idx >= IMG_ELEMS) return;
    int xx = idx & 127, yy = (idx >> 7) & 127, cc = (idx >> 14) & 63, bb = idx >> 20;
    outh[(((size_t)(bb*HH + yy))*CH + cc)*WW + xx] = (_Float16)x[idx];
}

// ---------------------------------------------------------------------------
// WMMA fragment helpers (CDNA5 ISA 7.12.2 layouts, wave32)
// A 16x32 f16: lane m=l&15, half=l>>4; vgpr v holds K pair
//   K = (v&4?16:0) + 2*(v&3) + 8*half  (+k0).  B uses identical pattern with
//   "row" = N (i.e. B stored row-major [N][K] == column-major over K).
// ---------------------------------------------------------------------------
__device__ inline v16h load_frag16(const _Float16* base, int row0, int ld,
                                   int k0, int lane)
{
    int m = lane & 15, half = lane >> 4;
    const _Float16* r = base + (size_t)(row0 + m)*ld + k0 + 8*half;
    v16h f;
    #pragma unroll
    for (int v = 0; v < 8; ++v) {
        int kk = ((v & 4) ? 16 : 0) + 2*(v & 3);
        f[2*v]   = r[kk];
        f[2*v+1] = r[kk+1];
    }
    return f;
}

__device__ inline void store_d_f32(float* base, int row0, int col0, int ld,
                                   int lane, v8f d)
{
    int nn = lane & 15, half = lane >> 4;
    #pragma unroll
    for (int v = 0; v < 8; ++v)
        base[(size_t)(row0 + v + 8*half)*ld + col0 + nn] = d[v];
}

// ---------------------------------------------------------------------------
// Fused per-row attention: score GEMM (WMMA), dual softmax, m_relax validity,
// warp GEMMs (WMMA), tanh-gated blend.  One workgroup (8 wave32) per row n.
// LDS ~131KB: sA/sB 16K each, sS 64K f32, sP 32K f16, stats ~3K.
// ---------------------------------------------------------------------------
__global__ __launch_bounds__(256) void attn_kernel(
    const _Float16* __restrict__ Qh, const _Float16* __restrict__ Kh,
    const _Float16* __restrict__ xlh, const _Float16* __restrict__ xrh,
    const float* __restrict__ xL, const float* __restrict__ xR,
    float* __restrict__ outL, float* __restrict__ outR)
{
    __shared__ _Float16 sA[128*64];
    __shared__ _Float16 sB[128*64];
    __shared__ float    sS[128*128];
    __shared__ _Float16 sP[128*128];
    __shared__ float rmax[128], rinv[128], cmax[128], cinv[128];
    __shared__ float sVl[128], sVr[128];

    int n = blockIdx.x, bb = n >> 7, y = n & 127;
    int t = threadIdx.x, wave = t >> 5, lane = t & 31;

    { // load Q/K rows (f16, layout [u][c]) as flat u32 copies
        const uint32_t* gq = (const uint32_t*)(Qh + (size_t)n*128*64);
        const uint32_t* gk = (const uint32_t*)(Kh + (size_t)n*128*64);
        uint32_t* lq = (uint32_t*)sA; uint32_t* lk = (uint32_t*)sB;
        for (int i = t; i < 4096; i += 256) { lq[i] = gq[i]; lk[i] = gk[i]; }
    }
    __syncthreads();

    // ---- GEMM1: S[u,v] = sum_c Q[u,c]*K[v,c] : wave handles S tile-row tm
    {
        int tm = wave;
        v16h a0 = load_frag16(sA, tm*16, 64, 0,  lane);
        v16h a1 = load_frag16(sA, tm*16, 64, 32, lane);
        for (int tn = 0; tn < 8; ++tn) {
            v16h b0 = load_frag16(sB, tn*16, 64, 0,  lane);
            v16h b1 = load_frag16(sB, tn*16, 64, 32, lane);
            v8f c = {};
            c = __builtin_amdgcn_wmma_f32_16x16x32_f16(false, a0, false, b0,
                                                       (short)0, c, false, false);
            c = __builtin_amdgcn_wmma_f32_16x16x32_f16(false, a1, false, b1,
                                                       (short)0, c, false, false);
            store_d_f32(sS, tm*16, tn*16, 128, lane, c);
        }
    }
    __syncthreads();

    // ---- dual softmax stats (rows for M_rl, columns for M_lr)
    if (t < 128) {
        float m = -3.4e38f;
        for (int j = 0; j < 128; ++j) m = fmaxf(m, sS[t*128+j]);
        float s = 0.f;
        for (int j = 0; j < 128; ++j) s += __expf(sS[t*128+j] - m);
        rmax[t] = m; rinv[t] = 1.f / s;
    } else {
        int col = t - 128;
        float m = -3.4e38f;
        for (int i = 0; i < 128; ++i) m = fmaxf(m, sS[i*128+col]);
        float s = 0.f;
        for (int i = 0; i < 128; ++i) s += __expf(sS[i*128+col] - m);
        cmax[col] = m; cinv[col] = 1.f / s;
    }
    __syncthreads();

    // ---- materialize P = M_rl (f16)
    for (int i = t; i < 128*128; i += 256) {
        int u = i >> 7;
        sP[i] = (_Float16)(__expf(sS[i] - rmax[u]) * rinv[u]);
    }
    __syncthreads();

    // ---- validity maps: V = sum_v m_relax(M)[u,v] * M'[v,u], then tanh(5V)
    if (t < 128) {
        int u = t; float acc = 0.f;
        for (int v = 0; v < 128; ++v) {
            float rel = 0.f;
            #pragma unroll
            for (int i = -2; i <= 2; ++i) {
                int uu = u + i;
                if (uu >= 0 && uu < 128) rel += (float)sP[uu*128 + v];
            }
            acc += rel * __expf(sS[u*128+v] - cmax[v]) * cinv[v];   // M_lr[v,u]
        }
        sVl[u] = tanhf(5.f*acc);
    } else {
        int u = t - 128; float acc = 0.f;
        for (int v = 0; v < 128; ++v) {
            float rel = 0.f;
            #pragma unroll
            for (int i = -2; i <= 2; ++i) {
                int uu = u + i;
                if (uu >= 0 && uu < 128)
                    rel += __expf(sS[v*128+uu] - cmax[uu]) * cinv[uu]; // M_lr[u+i,v]
            }
            acc += rel * (float)sP[v*128 + u];                         // M_rl[v,u]
        }
        sVr[u] = tanhf(5.f*acc);
    }
    __syncthreads();

    { // reuse sA/sB for transposed f16 images [c][v]
        const uint32_t* gr = (const uint32_t*)(xrh + (size_t)n*64*128);
        const uint32_t* gl = (const uint32_t*)(xlh + (size_t)n*64*128);
        uint32_t* la = (uint32_t*)sA; uint32_t* lb = (uint32_t*)sB;
        for (int i = t; i < 4096; i += 256) { la[i] = gr[i]; lb[i] = gl[i]; }
    }
    __syncthreads();

    // ---- GEMM2a: out_leftT[c,u] = sum_v xr[c,v] * M_rl[u,v]; blend + store
    {
        int tm = wave & 3, tnb = (wave >> 2) * 4;
        v16h a[4];
        #pragma unroll
        for (int ks = 0; ks < 4; ++ks) a[ks] = load_frag16(sA, tm*16, 128, ks*32, lane);
        for (int j = 0; j < 4; ++j) {
            int tn = tnb + j;
            v8f c = {};
            #pragma unroll
            for (int ks = 0; ks < 4; ++ks) {
                v16h b = load_frag16(sP, tn*16, 128, ks*32, lane);
                c = __builtin_amdgcn_wmma_f32_16x16x32_f16(false, a[ks], false, b,
                                                           (short)0, c, false, false);
            }
            int nn = lane & 15, half = lane >> 4;
            int u = tn*16 + nn;
            float vl = sVl[u];
            #pragma unroll
            for (int v = 0; v < 8; ++v) {
                int cch = tm*16 + v + 8*half;
                size_t gi = ((((size_t)bb*CH + cch)*HH + y))*WW + u;
                outL[gi] = xL[gi]*(1.f - vl) + c[v]*vl;
            }
        }
    }
    __syncthreads();

    // ---- overwrite P with M_lr: P[u][v] = exp(S[v,u]-cmax[u])/csum[u]
    for (int i = t; i < 128*128; i += 256) {
        int u = i >> 7, v = i & 127;
        sP[i] = (_Float16)(__expf(sS[v*128+u] - cmax[u]) * cinv[u]);
    }
    __syncthreads();

    // ---- GEMM2b: out_rightT[c,u] = sum_v xl[c,v] * M_lr[u,v]; blend + store
    {
        int tm = wave & 3, tnb = (wave >> 2) * 4;
        v16h a[4];
        #pragma unroll
        for (int ks = 0; ks < 4; ++ks) a[ks] = load_frag16(sB, tm*16, 128, ks*32, lane);
        for (int j = 0; j < 4; ++j) {
            int tn = tnb + j;
            v8f c = {};
            #pragma unroll
            for (int ks = 0; ks < 4; ++ks) {
                v16h b = load_frag16(sP, tn*16, 128, ks*32, lane);
                c = __builtin_amdgcn_wmma_f32_16x16x32_f16(false, a[ks], false, b,
                                                           (short)0, c, false, false);
            }
            int nn = lane & 15, half = lane >> 4;
            int u = tn*16 + nn;
            float vr = sVr[u];
            #pragma unroll
            for (int v = 0; v < 8; ++v) {
                int cch = tm*16 + v + 8*half;
                size_t gi = ((((size_t)bb*CH + cch)*HH + y))*WW + u;
                outR[gi] = xR[gi]*(1.f - vr) + c[v]*vr;
            }
        }
    }
}

// ---------------------------------------------------------------------------
extern "C" void kernel_launch(void* const* d_in, const int* in_sizes, int n_in,
                              void* d_out, int out_size, void* d_ws, size_t ws_size,
                              hipStream_t stream)
{
    const float* xl    = (const float*)d_in[0];
    const float* xr    = (const float*)d_in[1];
    const float* gamma = (const float*)d_in[2];
    const float* beta  = (const float*)d_in[3];
    const float* rb_w1 = (const float*)d_in[4];
    const float* rb_b1 = (const float*)d_in[5];
    const float* rb_w2 = (const float*)d_in[6];
    const float* rb_b2 = (const float*)d_in[7];
    const float* q_w   = (const float*)d_in[8];
    const float* q_b   = (const float*)d_in[9];
    const float* k_w   = (const float*)d_in[10];
    const float* k_b   = (const float*)d_in[11];

    float* outL = (float*)d_out;
    float* outR = outL + IMG_ELEMS;

    char* ws = (char*)d_ws;
    float*    stats = (float*)ws;                          // 1024 B
    float*    y1    = (float*)(ws + 1024);                 // reused L then R
    float*    resbL = (float*)(ws + 1024 + IMG_BYTES);
    float*    resbR = (float*)(ws + 1024 + 2*IMG_BYTES);
    _Float16* Qh    = (_Float16*)(ws + 1024 + 3*IMG_BYTES);
    _Float16* Kh    = Qh  + IMG_ELEMS;
    _Float16* xlh   = Kh  + IMG_ELEMS;
    _Float16* xrh   = xlh + IMG_ELEMS;

    bn_stats_kernel<<<dim3(CH, 2), 256, 0, stream>>>(xl, xr, stats);

    dim3 cgrid(64, 4, BATCH);
    // left: conv1 -> y1, conv2+residual -> resbL
    conv3x3_kernel<<<cgrid, 256, 0, stream>>>(xl, y1, rb_w1, rb_b1, stats,
                                              gamma, beta, y1,    0, 0);
    conv3x3_kernel<<<cgrid, 256, 0, stream>>>(xl, y1, rb_w2, rb_b2, stats,
                                              gamma, beta, resbL, 0, 1);
    // right
    conv3x3_kernel<<<cgrid, 256, 0, stream>>>(xr, y1, rb_w1, rb_b1, stats,
                                              gamma, beta, y1,    1, 0);
    conv3x3_kernel<<<cgrid, 256, 0, stream>>>(xr, y1, rb_w2, rb_b2, stats,
                                              gamma, beta, resbR, 1, 1);

    dim3 qgrid(CH, HH, BATCH);
    qk_kernel<<<qgrid, 128, 0, stream>>>(resbL, q_w, q_b, Qh);
    qk_kernel<<<qgrid, 128, 0, stream>>>(resbR, k_w, k_b, Kh);

    int tblocks = (int)((IMG_ELEMS + 255) / 256);
    tof16t_kernel<<<tblocks, 256, 0, stream>>>(xl, xlh);
    tof16t_kernel<<<tblocks, 256, 0, stream>>>(xr, xrh);

    attn_kernel<<<NROWS, 256, 0, stream>>>(Qh, Kh, xlh, xrh, xl, xr, outL, outR);
}
